// SelfAttention_38242388803679
// MI455X (gfx1250) — matrix operand order repaired and checked by reading, not compiled, not run on hardware
//
#include <hip/hip_runtime.h>

// Problem constants (match reference: B=4, C=512, H=W=64 -> N=4096)
#define BB 4
#define CC 512
#define NN 4096
#define QT 64   // queries (j) per workgroup
#define KT 64   // keys (i) per iteration

typedef __attribute__((ext_vector_type(16))) __bf16 v16bf;
typedef __attribute__((ext_vector_type(8)))  __bf16 bf16x8;
typedef __attribute__((ext_vector_type(4)))  __bf16 bf16x4;
typedef __attribute__((ext_vector_type(8)))  float  v8f;
typedef __attribute__((ext_vector_type(4)))  float  v4f;

// Packed f32x4 -> bf16x4 (fptrunc, round-to-nearest-even); lets the backend
// use packed v_cvt instructions instead of manual bit-twiddling.
__device__ inline bf16x4 cvt4(float a, float b, float c, float d) {
    v4f x; x[0] = a; x[1] = b; x[2] = c; x[3] = d;
    return __builtin_convertvector(x, bf16x4);
}

__device__ inline v16bf cat8(bf16x8 lo, bf16x8 hi) {
    return __builtin_shufflevector(lo, hi, 0,1,2,3,4,5,6,7,8,9,10,11,12,13,14,15);
}

// A-fragment (16x32 bf16) from a K-innermost (row-major, ld = row stride) source.
// CDNA5 layout: lane holds row m=lane&15; halves 0..7 = k=kh..kh+7,
// halves 8..15 = k=16+kh..16+kh+7, kh = 8*(lane>=16).  => 2x ds_load_b128.
__device__ inline v16bf load_a_frag(const __bf16* base, int ld, int lane) {
    const int m  = lane & 15;
    const int kh = (lane >> 4) << 3;
    const __bf16* p = base + m * ld + kh;
    return cat8(*(const bf16x8*)(p), *(const bf16x8*)(p + 16));
}

// B-fragment (32x16 bf16) from a K-innermost source (column n has stride ld).
// CDNA5 layout: lane holds column n=lane&15; halves 0..7 = k=koff..koff+7,
// halves 8..15 = k=16+koff..16+koff+7, koff = 8*(lane>=16).  => 2x ds_load_b128.
__device__ inline v16bf load_b_frag(const __bf16* base, int ld, int lane) {
    const int n    = lane & 15;
    const int koff = (lane >> 4) << 3;
    const __bf16* p = base + n * ld + koff;
    return cat8(*(const bf16x8*)(p), *(const bf16x8*)(p + 16));
}

// Stage a [CC x 64] f32 tile (row stride NN, starting column col0) into LDS
// row-major bf16 [CC][64]  (K = column index innermost).
__device__ inline void load_tile_rm(const float* __restrict__ g, __bf16* s, int col0) {
    for (int idx = threadIdx.x; idx < CC * 16; idx += 256) {
        const int c  = idx >> 4;
        const int f4 = idx & 15;
        const float4 d = *(const float4*)(g + (size_t)c * NN + col0 + f4 * 4);
        *(bf16x4*)(s + c * 64 + f4 * 4) = cvt4(d.x, d.y, d.z, d.w);
    }
}

// Stage a [CC x 64] f32 tile into LDS *transposed* bf16 [64][CC]
// (K = channel index innermost). 4x4 in-thread transpose blocks keep the
// global side float4-coalesced and the LDS side packed b64 stores.
__device__ inline void load_tile_tr(const float* __restrict__ g, __bf16* sT, int col0) {
    for (int idx = threadIdx.x; idx < (CC / 4) * 16; idx += 256) {
        const int cg = idx >> 4;           // channel block (4 rows)
        const int ig = idx & 15;           // column block (4 cols)
        const int c0 = cg * 4, i0 = ig * 4;
        const float4 r0 = *(const float4*)(g + (size_t)(c0 + 0) * NN + col0 + i0);
        const float4 r1 = *(const float4*)(g + (size_t)(c0 + 1) * NN + col0 + i0);
        const float4 r2 = *(const float4*)(g + (size_t)(c0 + 2) * NN + col0 + i0);
        const float4 r3 = *(const float4*)(g + (size_t)(c0 + 3) * NN + col0 + i0);
        *(bf16x4*)(sT + (i0 + 0) * CC + c0) = cvt4(r0.x, r1.x, r2.x, r3.x);
        *(bf16x4*)(sT + (i0 + 1) * CC + c0) = cvt4(r0.y, r1.y, r2.y, r3.y);
        *(bf16x4*)(sT + (i0 + 2) * CC + c0) = cvt4(r0.z, r1.z, r2.z, r3.z);
        *(bf16x4*)(sT + (i0 + 3) * CC + c0) = cvt4(r0.w, r1.w, r2.w, r3.w);
    }
}

__global__ __launch_bounds__(256)
void SelfAttention_fused_wmma(const float* __restrict__ q,
                              const float* __restrict__ k,
                              const float* __restrict__ v,
                              float* __restrict__ out) {
    extern __shared__ char smem[];
    __bf16* sQt = (__bf16*)smem;             // [64 j ][512 c] = 64 KB (resident)
    __bf16* sKt = sQt + 64 * CC;             // [64 i ][512 c] = 64 KB
    __bf16* sV  = sKt + 64 * CC;             // [512 c][64 i ] = 64 KB
    float*  sS  = (float*)(sV + CC * 64);    // [64 j ][64 i ] f32 scores = 16 KB
    __bf16* sPt = (__bf16*)(sS + 64 * 64);   // [64 j ][64 i ] bf16 probs = 8 KB
    float*  sM  = (float*)(sPt + 64 * 64);   // running max  [64]
    float*  sL  = sM + 64;                   // running sum  [64]
    float*  sSc = sL + 64;                   // rescale      [64]

    const int b    = blockIdx.y;
    const int jb   = blockIdx.x * QT;        // global query-column base
    const int tid  = threadIdx.x;
    const int lane = tid & 31;
    const int wave = tid >> 5;

    const size_t bo = (size_t)b * CC * NN;
    const float* qb = q + bo;
    const float* kb = k + bo;
    const float* vb = v + bo;

    // Phase-A: wave -> (i-row tile, pair of j tiles) of the 64x64 score tile
    const int itr = (wave & 3) * 16;
    const int jt0 = (wave >> 2) * 2;

    // Phase-B: wave owns 64 channels of the O accumulator [512 x 64]
    const int cb = wave * 64;
    v8f o[4][4] = {};

    load_tile_tr(qb, sQt, jb);
    if (tid < 64) { sM[tid] = -__builtin_inff(); sL[tid] = 0.f; }
    __syncthreads();

    for (int kt = 0; kt < NN / KT; ++kt) {
        const int ib = kt * KT;
        load_tile_tr(kb, sKt, ib);
        load_tile_rm(vb, sV, ib);
        __syncthreads();

        // ---- GEMM1: S(64x64) = K_tile^T * Q_tile, all fragments 2x b128 loads
        v8f s0 = {}, s1 = {};
#pragma unroll 4
        for (int kc = 0; kc < 16; ++kc) {
            v16bf a  = load_a_frag(sKt + itr * CC + kc * 32, CC, lane);
            v16bf b0 = load_b_frag(sQt + (jt0 * 16) * CC + kc * 32, CC, lane);
            v16bf b1 = load_b_frag(sQt + ((jt0 + 1) * 16) * CC + kc * 32, CC, lane);
            s0 = __builtin_amdgcn_wmma_f32_16x16x32_bf16(false, a, false, b0, (short)0, s0, false, false);
            s1 = __builtin_amdgcn_wmma_f32_16x16x32_bf16(false, a, false, b1, (short)0, s1, false, false);
        }
        {
            // C-tile: lane column j fixed, rows i = row..row+7 across the 8 VGPRs
            // -> two contiguous float4 stores into sS[j][i]
            const int row = itr + ((lane >> 4) << 3);
            const int c0  = jt0 * 16 + (lane & 15);
            float* p0 = sS + (c0)      * 64 + row;
            float* p1 = sS + (c0 + 16) * 64 + row;
            *(float4*)(p0)     = make_float4(s0[0], s0[1], s0[2], s0[3]);
            *(float4*)(p0 + 4) = make_float4(s0[4], s0[5], s0[6], s0[7]);
            *(float4*)(p1)     = make_float4(s1[0], s1[1], s1[2], s1[3]);
            *(float4*)(p1 + 4) = make_float4(s1[4], s1[5], s1[6], s1[7]);
        }
        __syncthreads();

        // ---- Online softmax over key index i (one thread per query column j);
        //      contiguous row reads, packed bf16x4 row writes.
        if (tid < 64) {
            const int j = tid;
            const float mold = sM[j];
            const float* srow = sS + j * 64;
            float tmax = -__builtin_inff();
            for (int i = 0; i < 64; ++i) tmax = fmaxf(tmax, srow[i]);
            const float mnew = fmaxf(mold, tmax);
            const float sc   = __expf(mold - mnew);   // 0 on first iteration
            float lsum = 0.f;
            for (int i0 = 0; i0 < 64; i0 += 4) {
                const float p0 = __expf(srow[i0 + 0] - mnew);
                const float p1 = __expf(srow[i0 + 1] - mnew);
                const float p2 = __expf(srow[i0 + 2] - mnew);
                const float p3 = __expf(srow[i0 + 3] - mnew);
                lsum += p0 + p1 + p2 + p3;
                *(bf16x4*)(sPt + j * 64 + i0) = cvt4(p0, p1, p2, p3);
            }
            sL[j]  = sL[j] * sc + lsum;
            sM[j]  = mnew;
            sSc[j] = sc;
        }
        __syncthreads();

        // ---- Rescale accumulators by exp(m_old - m_new) per query column
#pragma unroll
        for (int nt = 0; nt < 4; ++nt) {
            const float sc = sSc[nt * 16 + (lane & 15)];
#pragma unroll
            for (int mt = 0; mt < 4; ++mt)
#pragma unroll
                for (int vv = 0; vv < 8; ++vv) o[mt][nt][vv] *= sc;
        }

        // ---- GEMM2: O(64x64 per wave) += V_tile * P
#pragma unroll
        for (int kc = 0; kc < 2; ++kc) {
            v16bf bp[4];
#pragma unroll
            for (int nt = 0; nt < 4; ++nt)
                bp[nt] = load_b_frag(sPt + (nt * 16) * 64 + kc * 32, 64, lane);
#pragma unroll
            for (int mt = 0; mt < 4; ++mt) {
                v16bf a = load_a_frag(sV + (cb + mt * 16) * 64 + kc * 32, 64, lane);
#pragma unroll
                for (int nt = 0; nt < 4; ++nt)
                    o[mt][nt] = __builtin_amdgcn_wmma_f32_16x16x32_bf16(false, a, false, bp[nt], (short)0, o[mt][nt], false, false);
            }
        }
        __syncthreads();  // protect sKt/sV/sPt before next iteration overwrites
    }

    // ---- Epilogue: out = O / l + residual q (exact f32 residual from global)
    float* ob = out + bo;
#pragma unroll
    for (int nt = 0; nt < 4; ++nt) {
        const int j = jb + nt * 16 + (lane & 15);
        const float linv = 1.0f / sL[nt * 16 + (lane & 15)];
#pragma unroll
        for (int mt = 0; mt < 4; ++mt) {
            const int c0 = cb + mt * 16 + ((lane >> 4) << 3);
#pragma unroll
            for (int vv = 0; vv < 8; ++vv) {
                const size_t idx = (size_t)(c0 + vv) * NN + j;
                ob[idx] = o[mt][nt][vv] * linv + qb[idx];
            }
        }
    }
}

extern "C" void kernel_launch(void* const* d_in, const int* in_sizes, int n_in,
                              void* d_out, int out_size, void* d_ws, size_t ws_size,
                              hipStream_t stream) {
    const float* q = (const float*)d_in[0];
    const float* k = (const float*)d_in[1];
    const float* v = (const float*)d_in[2];
    float* out = (float*)d_out;
    (void)in_sizes; (void)n_in; (void)out_size; (void)d_ws; (void)ws_size;

    // LDS: 3 * 64KB bf16 tiles + 16KB scores + 8KB probs + 3*256B stats
    const size_t shmem = (size_t)(3 * CC * 64 * 2) + 64 * 64 * 4 + 64 * 64 * 2 + 3 * 64 * 4;

    dim3 grid(NN / QT, BB);
    SelfAttention_fused_wmma<<<grid, 256, shmem, stream>>>(q, k, v, out);
}
